// DCN__36636071034949
// MI455X (gfx1250) — matrix operand order repaired and checked by reading, not compiled, not run on hardware
//
#include <hip/hip_runtime.h>
#include <hip/hip_bf16.h>
#include <math.h>

// ---------------------------------------------------------------------------
// DCNv2 fused pipeline for MI455X (gfx1250, wave32, WMMA bf16 -> f32)
//   K ordering everywhere: K = k*256 + c   (kernel tap outer, channel inner)
// ---------------------------------------------------------------------------

#define CIN   256
#define COUT  256
#define K2    9
#define IH    128
#define IW    128
#define IHW   (IH * IW)          // 16384
#define NB    2
#define KDIM  (K2 * CIN)         // 2304
#define NPIX  (NB * IHW)         // 32768

typedef __attribute__((ext_vector_type(16))) __bf16 bf16x16;
typedef __attribute__((ext_vector_type(8)))  __bf16 bf16x8;
typedef __attribute__((ext_vector_type(8)))  float  f32x8;

// ---------------------------------------------------------------------------
// Kernel 0: repack weights to bf16, K = (k, c) ordering, N-major rows.
// ---------------------------------------------------------------------------
__global__ __launch_bounds__(256) void prep_weights(
    const float* __restrict__ w_off, const float* __restrict__ w_mask,
    const float* __restrict__ w_conv,
    __bf16* __restrict__ Wom, __bf16* __restrict__ Wc) {
  int idx = blockIdx.x * 256 + threadIdx.x;
  const int total = (32 + COUT) * KDIM;
  if (idx >= total) return;
  int row = idx / KDIM;
  int K   = idx - row * KDIM;
  int k   = K >> 8;          // tap (CIN == 256)
  int c   = K & 255;         // channel
  if (row < 32) {
    float v = 0.f;
    if (row < 18)       v = w_off [(row * CIN + c) * K2 + k];
    else if (row < 27)  v = w_mask[((row - 18) * CIN + c) * K2 + k];
    Wom[(size_t)row * KDIM + K] = (__bf16)v;
  } else {
    int o = row - 32;
    Wc[(size_t)o * KDIM + K] = (__bf16)w_conv[(o * CIN + c) * K2 + k];
  }
}

// ---------------------------------------------------------------------------
// Kernel 1: offset/mask conv as im2col WMMA GEMM.
//   M-tile = 64 pixels, N = 32 (18 offset + 9 mask + 5 pad).
// ---------------------------------------------------------------------------
__global__ __launch_bounds__(256) void offset_mask_conv(
    const float* __restrict__ x, const __bf16* __restrict__ Wom,
    const float* __restrict__ b_off, const float* __restrict__ b_mask,
    float* __restrict__ offmask) {
  __shared__ __align__(16) __bf16 As[2][64 * 40];

  const int tid  = threadIdx.x;
  const int lane = tid & 31;
  const int wid  = tid >> 5;
  const int pix0 = blockIdx.x * 64;
  const int b    = pix0 >> 14;

  // staging mapping: thread -> (pixel row, 8-channel group)
  const int sm  = tid >> 2;                      // 0..63
  const int scg = tid & 3;                       // 0..3
  const int shw = (pix0 + sm) & (IHW - 1);
  const int sh  = shw >> 7, sw = shw & 127;
  const int sstore = sm * 40 + scg * 8;
  const float* xbase = x + (size_t)(b * CIN + scg * 8) * IHW;

  // wmma mapping
  const int mrow = wid >> 1;                     // 0..3
  const int n0   = (wid & 1) * 16;               // 0 or 16
  const int aoff = (mrow * 16 + (lane & 15)) * 40 + ((lane >> 4) << 4);
  f32x8 acc = {};

  int par = 0;
  #pragma unroll 1
  for (int k = 0; k < K2; ++k) {
    // ---- hoisted per-tap geometry ----
    const int ky = k / 3 - 1, kx = k - (k / 3) * 3 - 1;   // PAD = 1
    const int y = sh + ky, xx = sw + kx;
    const bool ok  = (y >= 0) & (y < IH) & (xx >= 0) & (xx < IW);
    const float okf = ok ? 1.f : 0.f;
    const int  yc  = min(max(y, 0), IH - 1);
    const int  xc  = min(max(xx, 0), IW - 1);
    const int  off = yc * IW + xc;                         // always valid

    #pragma unroll 1
    for (int cb = 0; cb < 8; ++cb) {
      const int c0 = cb * 32;
      // ---- stage A chunk: unconditional clamped loads ----
      {
        const float* xp = xbase + (size_t)c0 * IHW + off;
        union { __bf16 h[8]; uint4 v; } pk;
        #pragma unroll
        for (int j = 0; j < 8; ++j) {
          float v = xp[(size_t)j * IHW] * okf;
          pk.h[j] = (__bf16)v;
        }
        *reinterpret_cast<uint4*>(&As[par][sstore]) = pk.v;
      }
      __syncthreads();

      // ---- consume: A from LDS, B from global, 1 WMMA ----
      const __bf16* Ar = &As[par][aoff];
      bf16x8 alo = *reinterpret_cast<const bf16x8*>(Ar);
      bf16x8 ahi = *reinterpret_cast<const bf16x8*>(Ar + 8);
      bf16x16 a;
      #pragma unroll
      for (int i = 0; i < 8; ++i) { a[i] = alo[i]; a[8 + i] = ahi[i]; }

      const __bf16* Bp = Wom + (size_t)(n0 + (lane & 15)) * KDIM
                             + k * 256 + c0 + ((lane >> 4) << 4);
      bf16x16 bf = *reinterpret_cast<const bf16x16*>(Bp);
      __builtin_prefetch((const void*)(Bp + 32), 0, 3);  // next chunk's B

      acc = __builtin_amdgcn_wmma_f32_16x16x32_bf16(
          false, a, false, bf, (short)0, acc, false, false);
      par ^= 1;
    }
  }

  // epilogue: bias (+ fast sigmoid for mask channels), write offmask[pix][32]
  const int n = n0 + (lane & 15);
  float bias = 0.f;
  bool ismask = false;
  if (n < 18)      bias = b_off[n];
  else if (n < 27) { bias = b_mask[n - 18]; ismask = true; }
  #pragma unroll
  for (int i = 0; i < 8; ++i) {
    const int m = mrow * 16 + i + ((lane >> 4) << 3);
    float v = acc[i] + bias;
    if (ismask) v = __builtin_amdgcn_rcpf(1.f + __expf(-v));  // v_rcp_f32
    offmask[(size_t)(pix0 + m) * 32 + n] = v;
  }
}

// ---------------------------------------------------------------------------
// Kernel 2: fused deformable sampling + main GEMM.
//   M-tile = 64 pixels, N = 256 (full). Bilinear setup hoisted per tap;
//   validity folded into corner weights -> all gathers unconditional.
// ---------------------------------------------------------------------------
__global__ __launch_bounds__(256) void dcn_main(
    const float* __restrict__ x, const __bf16* __restrict__ Wc,
    const float* __restrict__ offm, const float* __restrict__ b_conv,
    float* __restrict__ out) {
  __shared__ __align__(16) __bf16 As[2][64 * 40];

  const int tid  = threadIdx.x;
  const int lane = tid & 31;
  const int wid  = tid >> 5;
  const int pix0 = blockIdx.x * 64;
  const int b    = pix0 >> 14;

  // staging mapping
  const int sm   = tid >> 2;
  const int scg  = tid & 3;
  const int spix = pix0 + sm;
  const int shw  = spix & (IHW - 1);
  const int sh   = shw >> 7, sw = shw & 127;
  const int sstore = sm * 40 + scg * 8;
  const float* xbase = x + (size_t)(b * CIN + scg * 8) * IHW;

  // wmma mapping
  const int mrow  = wid & 3;
  const int nhalf = wid >> 2;
  const int aoff  = (mrow * 16 + (lane & 15)) * 40 + ((lane >> 4) << 4);
  f32x8 acc[8];
  #pragma unroll
  for (int t = 0; t < 8; ++t) acc[t] = {};

  int par = 0;
  #pragma unroll 1
  for (int k = 0; k < K2; ++k) {
    // ---- hoisted per-tap bilinear setup (shared by all 256 channels) ----
    const float dy  = offm[(size_t)spix * 32 + 2 * k];
    const float dx  = offm[(size_t)spix * 32 + 2 * k + 1];
    const float msk = offm[(size_t)spix * 32 + 18 + k];
    const int   kyi = k / 3, kxi = k - kyi * 3;
    const float py  = (float)(sh - 1 + kyi) + dy;
    const float px  = (float)(sw - 1 + kxi) + dx;
    const float y0f = floorf(py), x0f = floorf(px);
    const int   y0  = (int)y0f,  x0 = (int)x0f;
    const float wy  = py - y0f,  wx = px - x0f;
    const bool iy0 = (y0 >= 0) & (y0 < IH);
    const bool iy1 = (y0 + 1 >= 0) & (y0 + 1 < IH);
    const bool ix0 = (x0 >= 0) & (x0 < IW);
    const bool ix1 = (x0 + 1 >= 0) & (x0 + 1 < IW);
    // validity folded into weights -> gathers are unconditional
    const float w00 = (iy0 & ix0) ? (1.f - wy) * (1.f - wx) * msk : 0.f;
    const float w01 = (iy0 & ix1) ? (1.f - wy) * wx * msk : 0.f;
    const float w10 = (iy1 & ix0) ? wy * (1.f - wx) * msk : 0.f;
    const float w11 = (iy1 & ix1) ? wy * wx * msk : 0.f;
    const int y0c = min(max(y0, 0), IH - 1), y1c = min(max(y0 + 1, 0), IH - 1);
    const int x0c = min(max(x0, 0), IW - 1), x1c = min(max(x0 + 1, 0), IW - 1);
    const int i00 = y0c * IW + x0c, i01 = y0c * IW + x1c;
    const int i10 = y1c * IW + x0c, i11 = y1c * IW + x1c;
    const int kc256 = k * 256;

    #pragma unroll 1
    for (int cb = 0; cb < 8; ++cb) {
      const int c0 = cb * 32;
      // ---- stage A chunk: 4 unconditional gathers per channel ----
      {
        const float* xp = xbase + (size_t)c0 * IHW;
        union { __bf16 h[8]; uint4 v; } pk;
        #pragma unroll
        for (int j = 0; j < 8; ++j) {
          const float* p = xp + (size_t)j * IHW;     // L2-resident gathers
          float v = w00 * p[i00];
          v = fmaf(w01, p[i01], v);
          v = fmaf(w10, p[i10], v);
          v = fmaf(w11, p[i11], v);
          pk.h[j] = (__bf16)v;
        }
        *reinterpret_cast<uint4*>(&As[par][sstore]) = pk.v;
      }
      __syncthreads();

      // ---- consume: A fragment reused across 8 N-tiles ----
      const __bf16* Ar = &As[par][aoff];
      bf16x8 alo = *reinterpret_cast<const bf16x8*>(Ar);
      bf16x8 ahi = *reinterpret_cast<const bf16x8*>(Ar + 8);
      bf16x16 a;
      #pragma unroll
      for (int i = 0; i < 8; ++i) { a[i] = alo[i]; a[8 + i] = ahi[i]; }

      const int kc = kc256 + c0 + ((lane >> 4) << 4);
      #pragma unroll
      for (int t = 0; t < 8; ++t) {
        const int n = nhalf * 128 + t * 16 + (lane & 15);
        const __bf16* Bp = Wc + (size_t)n * KDIM + kc;
        bf16x16 bf = *reinterpret_cast<const bf16x16*>(Bp);
        // pull next chunk's B line into L0 (K offset is always +32, even
        // across the tap boundary since c0+32 wraps into (k+1)*256)
        __builtin_prefetch((const void*)(Bp + 32), 0, 3);
        acc[t] = __builtin_amdgcn_wmma_f32_16x16x32_bf16(
            false, a, false, bf, (short)0, acc[t], false, false);
      }
      par ^= 1;
    }
  }

  // epilogue: + b_conv, store NCHW (consecutive M -> consecutive addresses)
  const int mbase = mrow * 16 + ((lane >> 4) << 3);
  const int ploc  = (pix0 & (IHW - 1)) + mbase;
  #pragma unroll
  for (int t = 0; t < 8; ++t) {
    const int n = nhalf * 128 + t * 16 + (lane & 15);
    const float bias = b_conv[n];
    float* op = out + (size_t)(b * COUT + n) * IHW + ploc;
    float4 lo = make_float4(acc[t][0] + bias, acc[t][1] + bias,
                            acc[t][2] + bias, acc[t][3] + bias);
    float4 hi = make_float4(acc[t][4] + bias, acc[t][5] + bias,
                            acc[t][6] + bias, acc[t][7] + bias);
    *reinterpret_cast<float4*>(op)     = lo;
    *reinterpret_cast<float4*>(op + 4) = hi;
  }
}

// ---------------------------------------------------------------------------
extern "C" void kernel_launch(void* const* d_in, const int* in_sizes, int n_in,
                              void* d_out, int out_size, void* d_ws, size_t ws_size,
                              hipStream_t stream) {
  (void)in_sizes; (void)n_in; (void)out_size; (void)ws_size;
  const float* x      = (const float*)d_in[0];
  const float* w_conv = (const float*)d_in[1];
  const float* b_conv = (const float*)d_in[2];
  const float* w_off  = (const float*)d_in[3];
  const float* b_off  = (const float*)d_in[4];
  const float* w_mask = (const float*)d_in[5];
  const float* b_mask = (const float*)d_in[6];
  float* out = (float*)d_out;

  // workspace layout (all 256B-aligned by construction)
  char* ws = (char*)d_ws;
  __bf16* Wom = (__bf16*)(ws);                                  //  32*2304 bf16
  __bf16* Wc  = (__bf16*)(ws + (size_t)32 * KDIM * 2);          // 256*2304 bf16
  float*  offm = (float*)(ws + (size_t)(32 + COUT) * KDIM * 2); // 32768*32 f32

  const int prep_total = (32 + COUT) * KDIM;
  prep_weights<<<(prep_total + 255) / 256, 256, 0, stream>>>(
      w_off, w_mask, w_conv, Wom, Wc);
  offset_mask_conv<<<NPIX / 64, 256, 0, stream>>>(x, Wom, b_off, b_mask, offm);
  dcn_main<<<NPIX / 64, 256, 0, stream>>>(x, Wc, offm, b_conv, out);
}
